// CausalSelfAttention_58634893525726
// MI455X (gfx1250) — matrix operand order repaired and checked by reading, not compiled, not run on hardware
//
#include <hip/hip_runtime.h>
#include <cstdint>

// ---------------------------------------------------------------------------
// CDNA5 (gfx1250) causal self-attention: bf16 WMMA + async global->LDS tile
// staging with double buffering.   B=2, T=2048, C=1024, H=16, D=64
// ---------------------------------------------------------------------------

typedef __attribute__((ext_vector_type(16))) __bf16 v16bf;
typedef __attribute__((ext_vector_type(8)))  float  v8f;
typedef int v4i_vs __attribute__((vector_size(16)));   // matches builtin proto

#define B_  2
#define T_  2048
#define C_  1024
#define H_  16
#define D_  64

#if defined(__AMDGCN__) && __has_builtin(__builtin_amdgcn_global_load_async_to_lds_b128) && \
    __has_builtin(__builtin_amdgcn_s_wait_asynccnt)
#define USE_ASYNC 1
#else
#define USE_ASYNC 0
#endif

// ---- helpers ---------------------------------------------------------------

static __device__ __forceinline__ uint16_t f2bf(float f) {
    uint32_t u = __builtin_bit_cast(uint32_t, f);
    u += 0x7FFFu + ((u >> 16) & 1u);   // round-to-nearest-even
    return (uint16_t)(u >> 16);
}

// Build a v16bf fragment from two 16-byte LDS chunks (ds_load_b128 x2).
static __device__ __forceinline__ v16bf ldfrag(const uint16_t* p0, const uint16_t* p1) {
    union { uint4 u[2]; v16bf v; } x;
    x.u[0] = *(const uint4*)p0;
    x.u[1] = *(const uint4*)p1;
    return x.v;
}

static __device__ __forceinline__ v8f wmma_bf16(v16bf a, v16bf b, v8f c) {
    return __builtin_amdgcn_wmma_f32_16x16x32_bf16(false, a, false, b, (short)0, c,
                                                   false, false);
}

// 16-byte global -> LDS copy. Async (ASYNCcnt-tracked, no VGPR round trip)
// when the gfx1250 builtins are available; synchronous fallback otherwise.
static __device__ __forceinline__ void copy16_g2l(const uint16_t* g, uint16_t* l) {
#if USE_ASYNC
    auto gp = (__attribute__((address_space(1))) v4i_vs*)(uintptr_t)g;
    auto lp = (__attribute__((address_space(3))) v4i_vs*)(uint32_t)(uintptr_t)l;
    __builtin_amdgcn_global_load_async_to_lds_b128(gp, lp, 0, 0);
#else
    *(uint4*)l = *(const uint4*)g;
#endif
}

// ---- fp32 -> bf16 casts ----------------------------------------------------

__global__ void cast_f32_bf16(const float* __restrict__ src,
                              uint16_t* __restrict__ dst, int n) {
    for (int i = blockIdx.x * blockDim.x + threadIdx.x; i < n;
         i += gridDim.x * blockDim.x)
        dst[i] = f2bf(src[i]);
}

// src is [K][N] row-major f32; dst is [N][K] bf16 (transposed) so that GEMM
// B-tiles become contiguous 16B copies (async-LDS friendly, no LDS transpose).
__global__ void cast_transpose_f32_bf16(const float* __restrict__ src,
                                        uint16_t* __restrict__ dst,
                                        int K, int N) {
    for (int i = blockIdx.x * blockDim.x + threadIdx.x; i < K * N;
         i += gridDim.x * blockDim.x) {
        int k = i / N, n = i - k * N;
        dst[(size_t)n * K + k] = f2bf(src[i]);
    }
}

// ---- generic bf16 GEMM: C[M,N] = A[M,K] @ Bt[N,K]^T + bias -----------------
// 128x128 block tile, 8 waves (2x4), each wave 64x32 via 16x16x32 bf16 WMMA.
// Double-buffered LDS tiles staged by async global->LDS copies.
// mode 0: f32 output row-major.
// mode 1: scatter to Q[B,H,T,D], K[B,H,T,D], Vt[B,H,D,T] (bf16).

__global__ __launch_bounds__(256)
void gemm_bf16_kernel(const uint16_t* __restrict__ A,
                      const uint16_t* __restrict__ Bt,
                      const float*    __restrict__ bias,
                      float*          __restrict__ outF,
                      uint16_t*       __restrict__ Qo,
                      uint16_t*       __restrict__ Ko,
                      uint16_t*       __restrict__ Vo,
                      int M, int N, int K, int mode) {
    __shared__ __align__(16) uint16_t Asub[2][128 * 32];   // [m][k]
    __shared__ __align__(16) uint16_t Bsub[2][128 * 32];   // [n][k]

    const int tid  = threadIdx.x;
    const int lane = tid & 31;
    const int wave = tid >> 5;
    const int wm   = wave >> 2;          // 0..1
    const int wn   = wave & 3;           // 0..3
    const int hb   = lane >> 4;          // lane half
    const int l16  = lane & 15;
    const int bm   = blockIdx.y * 128;
    const int bn   = blockIdx.x * 128;

    v8f acc[4][2];
    const v8f zf = {0.f, 0.f, 0.f, 0.f, 0.f, 0.f, 0.f, 0.f};
#pragma unroll
    for (int mt = 0; mt < 4; mt++)
#pragma unroll
        for (int nt = 0; nt < 2; nt++) acc[mt][nt] = zf;

    // 4 async 16B copies per thread per stage (2 for A, 2 for B).
    auto stage = [&](int k0, int buf) {
#pragma unroll
        for (int i = 0; i < 2; i++) {
            int idx = tid + i * 256;           // [0,512)
            int r = idx >> 2, cu = idx & 3;    // row 0..127, 16B chunk 0..3
            copy16_g2l(A  + (size_t)(bm + r) * K + k0 + cu * 8, &Asub[buf][idx * 8]);
            copy16_g2l(Bt + (size_t)(bn + r) * K + k0 + cu * 8, &Bsub[buf][idx * 8]);
        }
    };

    const int nk = K / 32;
    stage(0, 0);
    for (int kt = 0; kt < nk; kt++) {
        const int cur = kt & 1;
        if (kt + 1 < nk) {
            stage((kt + 1) * 32, cur ^ 1);     // prefetch next tile
#if USE_ASYNC
            __builtin_amdgcn_s_wait_asynccnt(4);   // tile kt done (in-order)
#endif
        } else {
#if USE_ASYNC
            __builtin_amdgcn_s_wait_asynccnt(0);
#endif
        }
        __syncthreads();                       // all waves' copies visible

        v16bf bfr[2];
#pragma unroll
        for (int nt = 0; nt < 2; nt++) {
            const uint16_t* p = &Bsub[cur][(wn * 32 + nt * 16 + l16) * 32 + hb * 8];
            bfr[nt] = ldfrag(p, p + 16);
        }
#pragma unroll
        for (int mt = 0; mt < 4; mt++) {
            const uint16_t* p = &Asub[cur][(wm * 64 + mt * 16 + l16) * 32 + hb * 8];
            v16bf af = ldfrag(p, p + 16);
#pragma unroll
            for (int nt = 0; nt < 2; nt++)
                acc[mt][nt] = wmma_bf16(af, bfr[nt], acc[mt][nt]);
        }
        __syncthreads();                       // protect buffer reuse (WAR)
    }

    // epilogue: C layout -> row m = v + 8*half, col n = l16 (per 16x16 tile)
#pragma unroll
    for (int mt = 0; mt < 4; mt++) {
#pragma unroll
        for (int nt = 0; nt < 2; nt++) {
#pragma unroll
            for (int v = 0; v < 8; v++) {
                int row = bm + wm * 64 + mt * 16 + v + 8 * hb;
                int col = bn + wn * 32 + nt * 16 + l16;
                float val = acc[mt][nt][v] + bias[col];
                if (mode == 0) {
                    outF[(size_t)row * N + col] = val;
                } else {
                    int region = col >> 10;        // 0=q 1=k 2=v
                    int c = col & (C_ - 1);
                    int h = c >> 6, d = c & (D_ - 1);
                    int bb = row >> 11, t = row & (T_ - 1);
                    int bhh = bb * H_ + h;
                    uint16_t bv = f2bf(val);
                    if (region == 0)
                        Qo[((size_t)bhh * T_ + t) * D_ + d] = bv;
                    else if (region == 1)
                        Ko[((size_t)bhh * T_ + t) * D_ + d] = bv;
                    else
                        Vo[((size_t)bhh * D_ + d) * T_ + t] = bv;  // transposed
                }
            }
        }
    }
}

// ---- flash attention -------------------------------------------------------
// One block = 64 query rows of one (b,h); 4 waves, each wave owns 16 rows.
// Key/value chunks of 64 double-buffered in LDS via async copies;
// online softmax; O accumulated in f32 WMMA.

__global__ __launch_bounds__(128)
void attn_kernel(const uint16_t* __restrict__ Q,
                 const uint16_t* __restrict__ Kv,
                 const uint16_t* __restrict__ Vt,
                 uint16_t* __restrict__ Y) {
    __shared__ __align__(16) uint16_t Ksub[2][64 * 64];    // [key][d]
    __shared__ __align__(16) uint16_t Vsub[2][64 * 64];    // [d][key]
    __shared__ __align__(16) uint16_t Psub[4][16 * 64];    // per-wave [row][key]

    const int tid  = threadIdx.x;
    const int lane = tid & 31;
    const int w    = tid >> 5;
    const int hb   = lane >> 4;
    const int l16  = lane & 15;
    const int bh   = blockIdx.y;
    const int q0   = blockIdx.x * 64;
    const size_t base = (size_t)bh * T_ * D_;   // same stride for Q, K, Vt

    // Q fragments: rows q0+w*16+l16, d-chunks [0,32) and [32,64)
    v16bf a_q[2];
    {
        const uint16_t* qp = Q + base + (size_t)(q0 + w * 16 + l16) * D_ + hb * 8;
        a_q[0] = ldfrag(qp, qp + 16);
        a_q[1] = ldfrag(qp + 32, qp + 48);
    }

    float run_max[8], run_sum[8];
    v8f o[4];
    const v8f zf = {0.f, 0.f, 0.f, 0.f, 0.f, 0.f, 0.f, 0.f};
#pragma unroll
    for (int v = 0; v < 8; v++) { run_max[v] = -1e30f; run_sum[v] = 0.f; }
#pragma unroll
    for (int dt = 0; dt < 4; dt++) o[dt] = zf;

    const float scale = 0.125f;          // 1/sqrt(64)
    const int nchunk = blockIdx.x + 1;   // causal: keys up to q0+63

    // 8 async 16B copies per thread per stage (4 K + 4 V).
    auto stage_kv = [&](int j0, int buf) {
#pragma unroll
        for (int i = 0; i < 4; i++) {
            int idx = tid + i * 128;           // [0,512)
            int r = idx >> 3, cu = idx & 7;
            copy16_g2l(Kv + base + (size_t)(j0 + r) * D_ + cu * 8,
                       &Ksub[buf][idx * 8]);
            copy16_g2l(Vt + base + (size_t)r * T_ + j0 + cu * 8,
                       &Vsub[buf][idx * 8]);
        }
    };

    stage_kv(0, 0);
    for (int jc = 0; jc < nchunk; jc++) {
        const int j0  = jc * 64;
        const int cur = jc & 1;
        if (jc + 1 < nchunk) {
            stage_kv((jc + 1) * 64, cur ^ 1);
#if USE_ASYNC
            __builtin_amdgcn_s_wait_asynccnt(8);
#endif
        } else {
#if USE_ASYNC
            __builtin_amdgcn_s_wait_asynccnt(0);
#endif
        }
        __syncthreads();

        // S = Q * K^T  (16 rows x 64 keys per wave)
        v8f s[4];
#pragma unroll
        for (int nt = 0; nt < 4; nt++) s[nt] = zf;
#pragma unroll
        for (int nt = 0; nt < 4; nt++)
#pragma unroll
            for (int dc = 0; dc < 2; dc++) {
                const uint16_t* kp =
                    &Ksub[cur][(nt * 16 + l16) * 64 + dc * 32 + hb * 8];
                s[nt] = wmma_bf16(a_q[dc], ldfrag(kp, kp + 16), s[nt]);
            }

        // scale + causal mask + chunk row-max
        float cmax[8];
#pragma unroll
        for (int v = 0; v < 8; v++) cmax[v] = -1e30f;
#pragma unroll
        for (int nt = 0; nt < 4; nt++) {
            int key = j0 + nt * 16 + l16;
#pragma unroll
            for (int v = 0; v < 8; v++) {
                int qr = q0 + w * 16 + v + 8 * hb;
                float sv = s[nt][v] * scale;
                if (key > qr) sv = -1e30f;
                s[nt][v] = sv;
                cmax[v] = fmaxf(cmax[v], sv);
            }
        }
#pragma unroll
        for (int v = 0; v < 8; v++)
#pragma unroll
            for (int off = 1; off < 16; off <<= 1)
                cmax[v] = fmaxf(cmax[v], __shfl_xor(cmax[v], off, 32));

        // online softmax update
        float corr[8], csum[8];
#pragma unroll
        for (int v = 0; v < 8; v++) {
            float nm = fmaxf(run_max[v], cmax[v]);
            corr[v] = __expf(run_max[v] - nm);
            run_max[v] = nm;
            csum[v] = 0.f;
        }
#pragma unroll
        for (int nt = 0; nt < 4; nt++)
#pragma unroll
            for (int v = 0; v < 8; v++) {
                float p = __expf(s[nt][v] - run_max[v]);
                s[nt][v] = p;
                csum[v] += p;
            }
#pragma unroll
        for (int v = 0; v < 8; v++)
#pragma unroll
            for (int off = 1; off < 16; off <<= 1)
                csum[v] += __shfl_xor(csum[v], off, 32);
#pragma unroll
        for (int v = 0; v < 8; v++)
            run_sum[v] = run_sum[v] * corr[v] + csum[v];
#pragma unroll
        for (int dt = 0; dt < 4; dt++)
#pragma unroll
            for (int v = 0; v < 8; v++) o[dt][v] *= corr[v];

        // P -> per-wave LDS (C-layout -> A-layout transpose via LDS; DS ops
        // from one wave are processed in order, so no barrier is needed)
#pragma unroll
        for (int nt = 0; nt < 4; nt++)
#pragma unroll
            for (int v = 0; v < 8; v++)
                Psub[w][(v + 8 * hb) * 64 + nt * 16 + l16] = f2bf(s[nt][v]);

        v16bf a_p[2];
#pragma unroll
        for (int kc = 0; kc < 2; kc++) {
            const uint16_t* pp = &Psub[w][l16 * 64 + kc * 32 + hb * 8];
            a_p[kc] = ldfrag(pp, pp + 16);
        }
        // O += P * V
#pragma unroll
        for (int dt = 0; dt < 4; dt++)
#pragma unroll
            for (int kc = 0; kc < 2; kc++) {
                const uint16_t* vp =
                    &Vsub[cur][(dt * 16 + l16) * 64 + kc * 32 + hb * 8];
                o[dt] = wmma_bf16(a_p[kc], ldfrag(vp, vp + 16), o[dt]);
            }
        __syncthreads();
    }

    // normalize and store Y[b,t,h*64+d] as bf16
    const int b = bh >> 4, h = bh & 15;
    float inv[8];
#pragma unroll
    for (int v = 0; v < 8; v++) inv[v] = 1.0f / run_sum[v];
#pragma unroll
    for (int dt = 0; dt < 4; dt++)
#pragma unroll
        for (int v = 0; v < 8; v++) {
            int t   = q0 + w * 16 + v + 8 * hb;
            int col = h * D_ + dt * 16 + l16;
            Y[((size_t)(b * T_ + t)) * C_ + col] = f2bf(o[dt][v] * inv[v]);
        }
}

// ---- host launch -----------------------------------------------------------

extern "C" void kernel_launch(void* const* d_in, const int* in_sizes, int n_in,
                              void* d_out, int out_size, void* d_ws, size_t ws_size,
                              hipStream_t stream) {
    const float* x      = (const float*)d_in[0];
    const float* w_qkv  = (const float*)d_in[1];
    const float* b_qkv  = (const float*)d_in[2];
    const float* w_proj = (const float*)d_in[3];
    const float* b_proj = (const float*)d_in[4];
    float* out = (float*)d_out;

    uint8_t* ws = (uint8_t*)d_ws;
    const size_t nX  = (size_t)B_ * T_ * C_;        // 4 Mi
    const size_t nWq = (size_t)C_ * 3 * C_;         // 3 Mi
    const size_t nWp = (size_t)C_ * C_;             // 1 Mi
    const size_t nH  = (size_t)B_ * H_ * T_ * D_;   // 4 Mi (per Q/K/Vt)

    size_t off = 0;
    uint16_t* Xbf  = (uint16_t*)(ws + off); off += nX  * 2;
    uint16_t* WqT  = (uint16_t*)(ws + off); off += nWq * 2;   // [3C][C]
    uint16_t* WpT  = (uint16_t*)(ws + off); off += nWp * 2;   // [C][C]
    uint16_t* Qbf  = (uint16_t*)(ws + off); off += nH  * 2;
    uint16_t* Kbf  = (uint16_t*)(ws + off); off += nH  * 2;
    uint16_t* Vtbf = (uint16_t*)(ws + off); off += nH  * 2;
    uint16_t* Ybf  = (uint16_t*)(ws + off); off += nX  * 2;   // 48 MB total

    cast_f32_bf16<<<2048, 256, 0, stream>>>(x, Xbf, (int)nX);
    cast_transpose_f32_bf16<<<2048, 256, 0, stream>>>(w_qkv,  WqT, C_, 3 * C_);
    cast_transpose_f32_bf16<<<2048, 256, 0, stream>>>(w_proj, WpT, C_, C_);

    // qkv = x @ Wqkv + b  -> scatter Q, K, Vt
    dim3 g1(3 * C_ / 128, B_ * T_ / 128);            // (24, 32)
    gemm_bf16_kernel<<<g1, 256, 0, stream>>>(Xbf, WqT, b_qkv, nullptr,
                                             Qbf, Kbf, Vtbf,
                                             B_ * T_, 3 * C_, C_, 1);

    dim3 ga(T_ / 64, B_ * H_);                        // (32, 32)
    attn_kernel<<<ga, 128, 0, stream>>>(Qbf, Kbf, Vtbf, Ybf);

    // out = Y @ Wproj + b
    dim3 g2(C_ / 128, B_ * T_ / 128);                 // (8, 32)
    gemm_bf16_kernel<<<g2, 256, 0, stream>>>(Ybf, WpT, b_proj, out,
                                             nullptr, nullptr, nullptr,
                                             B_ * T_, C_, C_, 0);
}
